// RGAT_PyG_28776280883883
// MI455X (gfx1250) — matrix operand order repaired and checked by reading, not compiled, not run on hardware
//
#include <hip/hip_runtime.h>
#include <hip/hip_bf16.h>

#define NNODES 50000
#define NEDGES 250000
#define FIN    256
#define HD     256
#define NH     4
#define DH     64
#define NC     153

typedef __attribute__((ext_vector_type(16))) __bf16 v16bf;
typedef __attribute__((ext_vector_type(8)))  __bf16 v8bf;
typedef __attribute__((ext_vector_type(8)))  float  v8f;

// ---------------- small utility kernels ----------------

__global__ __launch_bounds__(256) void k_cast_bf16(const float* __restrict__ src,
                                                   __bf16* __restrict__ dst, int n) {
  int i = blockIdx.x * 256 + threadIdx.x;
  if (i < n) dst[i] = (__bf16)src[i];
}

__global__ __launch_bounds__(256) void k_fill(float* __restrict__ p, float v, int n) {
  int i = blockIdx.x * 256 + threadIdx.x;
  if (i < n) p[i] = v;
}

// Bt[n][k] = W[k][n]  (cast to bf16, zero-pad rows n >= ncols)
__global__ __launch_bounds__(256) void k_transpose_cast(const float* __restrict__ W,
                                                        __bf16* __restrict__ Bt,
                                                        int K, int ncols, int npad) {
  int i = blockIdx.x * 256 + threadIdx.x;
  if (i >= npad * K) return;
  int nrow = i / K, k = i - nrow * K;
  Bt[i] = (nrow < ncols) ? (__bf16)W[(size_t)k * ncols + nrow] : (__bf16)0.0f;
}

// acc[n][f] = b1[f] + b2[f]
__global__ __launch_bounds__(256) void k_init_acc_bias(float* __restrict__ acc,
                                                       const float* __restrict__ b1,
                                                       const float* __restrict__ b2) {
  int i = blockIdx.x * 256 + threadIdx.x;
  if (i >= NNODES * HD) return;
  int f = i & (HD - 1);
  acc[i] = b1[f] + b2[f];
}

// leaky-relu(slope) then cast to bf16 (slope=1.0 -> pure cast)
__global__ __launch_bounds__(256) void k_act_cast(const float* __restrict__ acc,
                                                  __bf16* __restrict__ xb, float slope) {
  int i = blockIdx.x * 256 + threadIdx.x;
  if (i >= NNODES * HD) return;
  float v = acc[i];
  v = (v > 0.0f) ? v : v * slope;
  xb[i] = (__bf16)v;
}

// ---------------- WMMA bf16 GEMM: C[M x ncols] = A[M x 256] * Bt[ncols x 256]^T ----------------
// Block = 256 threads = 8 waves in 2(row) x 4(col); block tile = 32 rows x 256 cols.
// Each wave: 16 rows x 64 cols = 4 WMMA f32 accumulators, K=256 fully unrolled (32 wmma/wave).

__global__ __launch_bounds__(256) void k_gemm_bf16(const __bf16* __restrict__ A,
                                                   const __bf16* __restrict__ Bt,
                                                   float* __restrict__ C,
                                                   int M, int ldc, int ncols,
                                                   const float* __restrict__ bias) {
  constexpr int K = 256;
  const int lane  = threadIdx.x & 31;
  const int wave  = threadIdx.x >> 5;
  const int wrow  = wave >> 2;                 // 0..1
  const int wcol  = wave & 3;                  // 0..3
  const int rowBase = blockIdx.x * 32 + wrow * 16;
  if (rowBase >= M) return;                    // wave-uniform exit (EXEC stays all-ones for WMMA)

  const int m16   = lane & 15;
  const int khalf = lane >> 4;                 // 0: lanes 0-15, 1: lanes 16-31

  v8f acc0 = {}, acc1 = {}, acc2 = {}, acc3 = {};

  // A fragment base: row = rowBase + (lane%16); contiguous K per lane half (matches B):
  // lanes 0-15 hold K {0..15}; lanes 16-31 hold K {16..31}.
  const __bf16* __restrict__ arow = A + (size_t)(rowBase + m16) * K;
  const __bf16* __restrict__ bcol0 = Bt + (size_t)(wcol * 64 + m16) * K + khalf * 16;

#pragma unroll
  for (int k0 = 0; k0 < K; k0 += 32) {
    // Build A fragment with contiguous K per lane half (same order as B).
    v8bf alo = *(const v8bf*)(arow + k0 + khalf * 16);
    v8bf ahi = *(const v8bf*)(arow + k0 + khalf * 16 + 8);
    v16bf afrag;
#pragma unroll
    for (int t = 0; t < 8; ++t) { afrag[t] = alo[t]; afrag[t + 8] = ahi[t]; }

    // B fragments: lane = output column, contiguous K per lane half (column-major weights).
    v16bf b0 = *(const v16bf*)(bcol0 + 0 * 16 * K + k0);
    v16bf b1 = *(const v16bf*)(bcol0 + 1 * 16 * K + k0);
    v16bf b2 = *(const v16bf*)(bcol0 + 2 * 16 * K + k0);
    v16bf b3 = *(const v16bf*)(bcol0 + 3 * 16 * K + k0);

    acc0 = __builtin_amdgcn_wmma_f32_16x16x32_bf16(false, afrag, false, b0, (short)0, acc0, false, false);
    acc1 = __builtin_amdgcn_wmma_f32_16x16x32_bf16(false, afrag, false, b1, (short)0, acc1, false, false);
    acc2 = __builtin_amdgcn_wmma_f32_16x16x32_bf16(false, afrag, false, b2, (short)0, acc2, false, false);
    acc3 = __builtin_amdgcn_wmma_f32_16x16x32_bf16(false, afrag, false, b3, (short)0, acc3, false, false);
  }

  // Store: C/D layout -> lane holds col = colBase + lane%16, rows v + 8*(lane>=16).
  v8f accs[4] = {acc0, acc1, acc2, acc3};
#pragma unroll
  for (int j = 0; j < 4; ++j) {
    int col = wcol * 64 + j * 16 + m16;
    if (col < ncols) {
      float badd = bias ? bias[col] : 0.0f;
#pragma unroll
      for (int v = 0; v < 8; ++v) {
        int row = rowBase + v + khalf * 8;
        C[(size_t)row * ldc + col] = accs[j][v] + badd;
      }
    }
  }
}

// ---------------- attention kernels ----------------

// s_src[n][h] = sum_d h[n][h*64+d]*a_src[h][d]; same for s_dst
__global__ __launch_bounds__(256) void k_node_scores(const float* __restrict__ h,
                                                     const float* __restrict__ asrc,
                                                     const float* __restrict__ adst,
                                                     float* __restrict__ ssrc,
                                                     float* __restrict__ sdst) {
  int tid = blockIdx.x * 256 + threadIdx.x;
  if (tid >= NNODES * NH) return;
  int n = tid >> 2, hh = tid & 3;
  const float* hp = h + (size_t)n * HD + hh * DH;
  const float* as = asrc + hh * DH;
  const float* ad = adst + hh * DH;
  float s1 = 0.0f, s2 = 0.0f;
#pragma unroll 8
  for (int d = 0; d < DH; ++d) { float v = hp[d]; s1 += v * as[d]; s2 += v * ad[d]; }
  ssrc[tid] = s1; sdst[tid] = s2;
}

__device__ inline void atomicMaxF(float* addr, float val) {
  // ordered-int trick: works for all finite floats
  if (val >= 0.0f) atomicMax((int*)addr, __float_as_int(val));
  else             atomicMin((unsigned int*)addr, __float_as_uint(val));
}

// e = leaky_relu(s_src[src]+s_dst[dst], 0.2); m[dst] = max(m[dst], e)
__global__ __launch_bounds__(256) void k_edge_scores(const int* __restrict__ ei,
                                                     const float* __restrict__ ssrc,
                                                     const float* __restrict__ sdst,
                                                     float* __restrict__ e,
                                                     float* __restrict__ m) {
  int tid = blockIdx.x * 256 + threadIdx.x;
  if (tid >= NEDGES * NH) return;
  int ed = tid >> 2, hh = tid & 3;
  int src = ei[ed], dst = ei[NEDGES + ed];
  float v = ssrc[src * NH + hh] + sdst[dst * NH + hh];
  v = (v > 0.0f) ? v : 0.2f * v;
  e[tid] = v;
  atomicMaxF(&m[dst * NH + hh], v);
}

// p = exp(e - m[dst]); z[dst] += p  (p stored in-place into e)
__global__ __launch_bounds__(256) void k_edge_exp(const int* __restrict__ ei,
                                                  float* __restrict__ e,
                                                  const float* __restrict__ m,
                                                  float* __restrict__ z) {
  int tid = blockIdx.x * 256 + threadIdx.x;
  if (tid >= NEDGES * NH) return;
  int ed = tid >> 2, hh = tid & 3;
  int dst = ei[NEDGES + ed];
  float p = __expf(e[tid] - m[dst * NH + hh]);
  e[tid] = p;
  atomicAdd(&z[dst * NH + hh], p);
}

// acc[dst][h*64+d] += h[src][h*64+d] * p/(z[dst]+1e-16)
__global__ __launch_bounds__(256) void k_edge_aggregate(const int* __restrict__ ei,
                                                        const float* __restrict__ p,
                                                        const float* __restrict__ z,
                                                        const float* __restrict__ h,
                                                        float* __restrict__ acc) {
  int tid = blockIdx.x * 256 + threadIdx.x;   // NEDGES*DH = 16M threads
  if (tid >= NEDGES * DH) return;
  int ed = tid >> 6, d = tid & 63;
  int src = ei[ed], dst = ei[NEDGES + ed];
  const float* hs = h + (size_t)src * HD;
  float* ao = acc + (size_t)dst * HD;
#pragma unroll
  for (int hh = 0; hh < NH; ++hh) {
    float alpha = p[ed * NH + hh] / (z[dst * NH + hh] + 1e-16f);
    atomicAdd(&ao[hh * DH + d], hs[hh * DH + d] * alpha);
  }
}

// ---------------- host orchestration ----------------

static void launch_etype(const __bf16* xb, const float* W, const float* asrc, const float* adst,
                         const int* ei, __bf16* wbt, float* hbuf, float* ssrc, float* sdst,
                         float* m, float* z, float* e, float* acc, hipStream_t s) {
  k_transpose_cast<<<(HD * HD + 255) / 256, 256, 0, s>>>(W, wbt, HD, HD, HD);
  k_gemm_bf16<<<(NNODES + 31) / 32, 256, 0, s>>>(xb, wbt, hbuf, NNODES, HD, HD, nullptr);
  k_node_scores<<<(NNODES * NH + 255) / 256, 256, 0, s>>>(hbuf, asrc, adst, ssrc, sdst);
  k_fill<<<(NNODES * NH + 255) / 256, 256, 0, s>>>(m, -1e30f, NNODES * NH);
  k_fill<<<(NNODES * NH + 255) / 256, 256, 0, s>>>(z, 0.0f, NNODES * NH);
  k_edge_scores<<<(NEDGES * NH + 255) / 256, 256, 0, s>>>(ei, ssrc, sdst, e, m);
  k_edge_exp<<<(NEDGES * NH + 255) / 256, 256, 0, s>>>(ei, e, m, z);
  k_edge_aggregate<<<(NEDGES * DH) / 256, 256, 0, s>>>(ei, e, z, hbuf, acc);
}

extern "C" void kernel_launch(void* const* d_in, const int* in_sizes, int n_in,
                              void* d_out, int out_size, void* d_ws, size_t ws_size,
                              hipStream_t stream) {
  (void)in_sizes; (void)n_in; (void)out_size; (void)ws_size;
  const float* x    = (const float*)d_in[0];
  const int*   ei_a = (const int*)d_in[1];
  const int*   ei_b = (const int*)d_in[2];
  const float* W0a  = (const float*)d_in[3];
  const float* as0a = (const float*)d_in[4];
  const float* ad0a = (const float*)d_in[5];
  const float* b0a  = (const float*)d_in[6];
  const float* W0b  = (const float*)d_in[7];
  const float* as0b = (const float*)d_in[8];
  const float* ad0b = (const float*)d_in[9];
  const float* b0b  = (const float*)d_in[10];
  const float* W1a  = (const float*)d_in[11];
  const float* as1a = (const float*)d_in[12];
  const float* ad1a = (const float*)d_in[13];
  const float* b1a  = (const float*)d_in[14];
  const float* W1b  = (const float*)d_in[15];
  const float* as1b = (const float*)d_in[16];
  const float* ad1b = (const float*)d_in[17];
  const float* b1b  = (const float*)d_in[18];
  const float* Wout = (const float*)d_in[19];
  const float* bout = (const float*)d_in[20];

  char* ws = (char*)d_ws;
  size_t off = 0;
  auto wsalloc = [&](size_t bytes) -> void* {
    off = (off + 255) & ~(size_t)255;
    void* p = ws + off;
    off += bytes;
    return p;
  };
  __bf16* xb   = (__bf16*)wsalloc((size_t)NNODES * HD * 2);
  __bf16* wbt  = (__bf16*)wsalloc((size_t)HD * HD * 2);
  float*  hbuf = (float*)wsalloc((size_t)NNODES * HD * 4);
  float*  acc  = (float*)wsalloc((size_t)NNODES * HD * 4);
  float*  ssrc = (float*)wsalloc((size_t)NNODES * NH * 4);
  float*  sdst = (float*)wsalloc((size_t)NNODES * NH * 4);
  float*  m    = (float*)wsalloc((size_t)NNODES * NH * 4);
  float*  z    = (float*)wsalloc((size_t)NNODES * NH * 4);
  float*  e    = (float*)wsalloc((size_t)NEDGES * NH * 4);

  // cast input features to bf16
  k_cast_bf16<<<(NNODES * FIN + 255) / 256, 256, 0, stream>>>(x, xb, NNODES * FIN);

  // ---- layer 0 ----
  k_init_acc_bias<<<(NNODES * HD + 255) / 256, 256, 0, stream>>>(acc, b0a, b0b);
  launch_etype(xb, W0a, as0a, ad0a, ei_a, wbt, hbuf, ssrc, sdst, m, z, e, acc, stream);
  launch_etype(xb, W0b, as0b, ad0b, ei_b, wbt, hbuf, ssrc, sdst, m, z, e, acc, stream);
  k_act_cast<<<(NNODES * HD + 255) / 256, 256, 0, stream>>>(acc, xb, 0.01f);

  // ---- layer 1 ----
  k_init_acc_bias<<<(NNODES * HD + 255) / 256, 256, 0, stream>>>(acc, b1a, b1b);
  launch_etype(xb, W1a, as1a, ad1a, ei_a, wbt, hbuf, ssrc, sdst, m, z, e, acc, stream);
  launch_etype(xb, W1b, as1b, ad1b, ei_b, wbt, hbuf, ssrc, sdst, m, z, e, acc, stream);
  k_act_cast<<<(NNODES * HD + 255) / 256, 256, 0, stream>>>(acc, xb, 1.0f);  // no activation, just cast

  // ---- output linear: d_out = h @ Wout + bout ----
  k_transpose_cast<<<(HD * HD + 255) / 256, 256, 0, stream>>>(Wout, wbt, HD, NC, HD);
  k_gemm_bf16<<<(NNODES + 31) / 32, 256, 0, stream>>>(xb, wbt, (float*)d_out, NNODES, NC, NC, bout);
}
